// ResidualVectorQuantizer_31568009626248
// MI455X (gfx1250) — matrix operand order repaired and checked by reading, not compiled, not run on hardware
//
#include <hip/hip_runtime.h>
#include <hip/hip_bf16.h>
#include <stdint.h>

// ---------------------------------------------------------------------------
// Residual VQ for MI455X (gfx1250, wave32, WMMA bf16, TDM async tiles).
// B=65536, C=1024, D=256, K=2048, 4 codebooks.
// ---------------------------------------------------------------------------

#define B_ROWS 65536
#define C_DIM  1024
#define D_DIM  256
#define K_CB   2048
#define NCB    4

typedef __attribute__((ext_vector_type(16))) __bf16 v16bf;
typedef __attribute__((ext_vector_type(8)))  float  v8f;
typedef __attribute__((ext_vector_type(4)))  unsigned int u32x4;
typedef __attribute__((ext_vector_type(8)))  int   i32x8;
typedef __attribute__((ext_vector_type(4)))  int   i32x4;

union FragBF {            // one WMMA bf16 operand: 16 halfwords per lane
  uint4 q[2];
  v16bf v;
};
union Q8 {                // 8 bf16 <-> 128-bit
  uint4  q;
  __bf16 e[8];
};

__device__ __forceinline__ __bf16 f2bf(float x) { return (__bf16)x; }

// ---------------------------------------------------------------------------
// TDM descriptor: DMA a 64 x 256 bf16 tile (row stride 256 elems) from global
// into LDS, with hardware padding of 4 DWORDs every 128 DWORDs so the LDS
// row stride becomes 512+16 bytes = [264] __bf16 -- matching fragment reads.
//   group0: count=1 | lds_addr | global_addr(57b) | type=2
//   group1: data_size=2B, pad_enable, pad_interval=128dw(code 6),
//           pad_amount=4dw(code 3); tensor_dim0=256, tensor_dim1=64,
//           tile_dim0=256, tile_dim1=64, tensor_dim0_stride=256
// Toolchain here exposes the 6-arg builtin:
//   (u32x4 g0, i32x8 g1, i32x4 g2, i32x4 g3, i32x8, i32 cpol)
// ---------------------------------------------------------------------------
__device__ __forceinline__ void tdm_load_cb_tile(const __bf16* gsrc,
                                                 unsigned lds_addr) {
  const unsigned long long ga = (unsigned long long)(uintptr_t)gsrc;
  u32x4 g0;
  g0[0] = 1u;                                                  // count=1
  g0[1] = lds_addr;                                            // LDS byte addr
  g0[2] = (unsigned)ga;                                        // gaddr[31:0]
  g0[3] = ((unsigned)(ga >> 32) & 0x01FFFFFFu) | (2u << 30);   // gaddr hi|type2
  i32x8 g1;
  g1[0] = 0x07910000;  // data_size=1(2B) | pad_en | interval=6 | amount=3
  g1[1] = 0x01000000;  // tensor_dim0 = 256 (low16 at bits [63:48])
  g1[2] = 0x00400000;  // tensor_dim1 = 64  (low16 at bits [95:80])
  g1[3] = 0x01000000;  // tile_dim0   = 256 (bits [127:112])
  g1[4] = 64;          // tile_dim1   = 64
  g1[5] = 256;         // tensor_dim0_stride = 256
  g1[6] = 0;
  g1[7] = 0;
  const i32x4 z4 = {0, 0, 0, 0};
  const i32x8 z8 = {0, 0, 0, 0, 0, 0, 0, 0};
  __builtin_amdgcn_tensor_load_to_lds(g0, g1, z4, z4, z8, 0);
}

// ---------------------------------------------------------------------------
// init: residual = z, z_q = 0, loss accumulator = 0   (float4 vectorized)
// ---------------------------------------------------------------------------
__global__ void rvq_init(const float* __restrict__ z, float* __restrict__ zq,
                         float* __restrict__ resid, float* __restrict__ acc) {
  size_t i = (size_t)blockIdx.x * blockDim.x + threadIdx.x;
  float4 v = ((const float4*)z)[i];
  ((float4*)resid)[i] = v;
  float4 zero = {0.f, 0.f, 0.f, 0.f};
  ((float4*)zq)[i] = zero;
  if (i == 0) *acc = 0.f;
}

// ---------------------------------------------------------------------------
// f32 -> bf16 conversion (weights / codebooks, done once per launch)
// ---------------------------------------------------------------------------
__global__ void f32_to_bf16(const float* __restrict__ s, __bf16* __restrict__ d,
                            int n) {
  int i = blockIdx.x * blockDim.x + threadIdx.x;
  if (i < n) d[i] = f2bf(s[i]);
}

// ---------------------------------------------------------------------------
// per-codebook-row squared norms: cbn[row] = sum_d cb[row,d]^2
// ---------------------------------------------------------------------------
__global__ void cb_norms(const float* __restrict__ cb, float* __restrict__ cbn) {
  __shared__ float red[256];
  int row = blockIdx.x, t = threadIdx.x;
  float v = cb[(size_t)row * D_DIM + t];
  red[t] = v * v;
  __syncthreads();
  for (int s = 128; s > 0; s >>= 1) {
    if (t < s) red[t] += red[t + s];
    __syncthreads();
  }
  if (t == 0) cbn[row] = red[0];
}

// ---------------------------------------------------------------------------
// Generic WMMA GEMM:  C[M,N] = A(f32)[M,Kdim] x B(bf16)[Kdim,N] + bias
//   mode 0 (IN):  outf[row*ldo+col] = v                  (z_e -> latents slab)
//   mode 1 (OUT): zq[o] += v; resid[o] -= v              (fused RVQ epilogue)
// Block = 256 threads = 8 waves; block tile 64x64; K step 32.
// ---------------------------------------------------------------------------
__launch_bounds__(256)
__global__ void rvq_gemm(const float* __restrict__ A, int lda, int Kdim,
                         const __bf16* __restrict__ Bm, int ldb,
                         const float* __restrict__ bias,
                         float* __restrict__ outf, int ldo,
                         float* __restrict__ zq, float* __restrict__ resid,
                         int mode) {
  __shared__ __align__(16) __bf16 As[64][40];   // 64 rows x 32 K (+pad)
  __shared__ __align__(16) __bf16 Bt[64][40];   // 64 cols x 32 K (transposed)

  const int tid = threadIdx.x;
  const int bm = blockIdx.y * 64;
  const int bn = blockIdx.x * 64;
  const int w = tid >> 5, lane = tid & 31;
  const int wm = w >> 1, wn = w & 1;
  const int lr = lane & 15, hi = lane >> 4;

  v8f c0 = {}, c1 = {};

  const int arow = tid >> 2, ac8 = (tid & 3) * 8;   // A: 4 thr/row, 8 f32 each
  const int brow = tid >> 3, bc8 = (tid & 7) * 8;   // B: 8 thr/row, 8 bf16 each

  for (int kb = 0; kb < Kdim; kb += 32) {
    // ---- stage A 64x32 (f32 -> bf16) ----
    const float* ap = A + (size_t)(bm + arow) * lda + kb + ac8;
    if (kb + 32 < Kdim) __builtin_prefetch(ap + 32, 0, 1);  // global_prefetch_b8
    float4 a0 = *(const float4*)ap;
    float4 a1 = *(const float4*)(ap + 4);
    Q8 qa;
    qa.e[0] = f2bf(a0.x); qa.e[1] = f2bf(a0.y);
    qa.e[2] = f2bf(a0.z); qa.e[3] = f2bf(a0.w);
    qa.e[4] = f2bf(a1.x); qa.e[5] = f2bf(a1.y);
    qa.e[6] = f2bf(a1.z); qa.e[7] = f2bf(a1.w);
    *(uint4*)&As[arow][ac8] = qa.q;

    // ---- stage B 32x64, transposed into Bt[col][k] ----
    Q8 qb;
    qb.q = *(const uint4*)(Bm + (size_t)(kb + brow) * ldb + bn + bc8);
#pragma unroll
    for (int j = 0; j < 8; ++j) Bt[bc8 + j][brow] = qb.e[j];
    __syncthreads();

    // ---- fragments per ISA 16-bit layouts ----
    FragBF fa;
    fa.q[0] = *(const uint4*)&As[wm * 16 + lr][hi * 8];        // K = hi*8 ..
    fa.q[1] = *(const uint4*)&As[wm * 16 + lr][16 + hi * 8];   // K = 16+hi*8 ..
    const int n0 = wn * 32 + lr;
    FragBF fb0, fb1;
    fb0.q[0] = *(const uint4*)&Bt[n0][hi * 8];
    fb0.q[1] = *(const uint4*)&Bt[n0][16 + hi * 8];
    fb1.q[0] = *(const uint4*)&Bt[n0 + 16][hi * 8];
    fb1.q[1] = *(const uint4*)&Bt[n0 + 16][16 + hi * 8];

    c0 = __builtin_amdgcn_wmma_f32_16x16x32_bf16(false, fa.v, false, fb0.v,
                                                 (short)0, c0, false, false);
    c1 = __builtin_amdgcn_wmma_f32_16x16x32_bf16(false, fa.v, false, fb1.v,
                                                 (short)0, c1, false, false);
    __syncthreads();
  }

  // ---- epilogue (C layout: VGPR r -> row hi*8+r, lane lr -> col) ----
  const int colg0 = bn + wn * 32 + lr;
  const int rowb = bm + wm * 16 + hi * 8;
  const float bias0 = bias[colg0];
  const float bias1 = bias[colg0 + 16];
#pragma unroll
  for (int r = 0; r < 8; ++r) {
    const int rowg = rowb + r;
    const float v0 = c0[r] + bias0;
    const float v1 = c1[r] + bias1;
    if (mode == 0) {
      outf[(size_t)rowg * ldo + colg0] = v0;
      outf[(size_t)rowg * ldo + colg0 + 16] = v1;
    } else {
      const size_t o0 = (size_t)rowg * ldo + colg0;
      const size_t o1 = o0 + 16;
      zq[o0] += v0;
      zq[o1] += v1;
      resid[o0] -= v0;
      resid[o1] -= v1;
    }
  }
}

// ---------------------------------------------------------------------------
// Distance + argmin with TDM double-buffered codebook tiles.
// d[b,k] = cbn[k] - 2 * (z_e[b] . cb[k])   (row-constant |z_e|^2 dropped).
// A tile (64x256) staged once; 32 chunks of 64 codebook rows, each chunk
// DMA'd by the Tensor Data Mover into the alternate LDS buffer while the
// current chunk is consumed by 16 WMMAs + the argmin scan.
// ---------------------------------------------------------------------------
__launch_bounds__(256)
__global__ void rvq_dist_argmin(const float* __restrict__ Ze, int lda,
                                const __bf16* __restrict__ cbb,
                                const float* __restrict__ cbn,
                                int* __restrict__ idxbuf,
                                float* __restrict__ codes, int step) {
  __shared__ __align__(16) __bf16 As[64][264];     // 64 rows x full K=256
  __shared__ __align__(16) __bf16 Bs[2][64][264];  // double-buffered cb tiles
  __shared__ float dbuf[64][68];
  __shared__ float redv[256];
  __shared__ int   redi[256];

  const int tid = threadIdx.x;
  const int bm = blockIdx.x * 64;
  const int w = tid >> 5, lane = tid & 31;
  const int wm = w >> 1, wn = w & 1;
  const int lr = lane & 15, hi = lane >> 4;

  const unsigned bs_lds0 = (unsigned)(uintptr_t)&Bs[0][0][0];
  const unsigned bs_lds1 = (unsigned)(uintptr_t)&Bs[1][0][0];

  // ---- kick off TDM DMA of chunk 0 while we stage A ----
  if (tid < 32) tdm_load_cb_tile(cbb, bs_lds0);

  // ---- stage A tile once: 64 x 256, f32 -> bf16 ----
  {
    const int row = tid >> 2;
    const int cbase = (tid & 3) * 64;
    const float* ap = Ze + (size_t)(bm + row) * lda + cbase;
#pragma unroll
    for (int i = 0; i < 8; ++i) {
      float4 a0 = *(const float4*)(ap + i * 8);
      float4 a1 = *(const float4*)(ap + i * 8 + 4);
      Q8 qa;
      qa.e[0] = f2bf(a0.x); qa.e[1] = f2bf(a0.y);
      qa.e[2] = f2bf(a0.z); qa.e[3] = f2bf(a0.w);
      qa.e[4] = f2bf(a1.x); qa.e[5] = f2bf(a1.y);
      qa.e[6] = f2bf(a1.z); qa.e[7] = f2bf(a1.w);
      *(uint4*)&As[row][cbase + i * 8] = qa.q;
    }
  }

  float bestv = 3.4e38f;
  int besti = 0;
  const int srow = tid >> 2, sc16 = (tid & 3) * 16;

  for (int nc = 0; nc < 32; ++nc) {
    const int cur = nc & 1;
    // wave 0: issue DMA for next chunk into alternate buffer, then wait for
    // the current chunk (TENSORcnt is in-order per wave: 1 outstanding = the
    // just-issued next chunk, so current tile is complete).
    if (tid < 32) {
      if (nc + 1 < 32) {
        tdm_load_cb_tile(cbb + (size_t)(nc + 1) * 64 * D_DIM,
                         cur ? bs_lds0 : bs_lds1);
        __builtin_amdgcn_s_wait_tensorcnt((short)1);
      } else {
        __builtin_amdgcn_s_wait_tensorcnt((short)0);
      }
    }
    __syncthreads();   // current tile visible to all waves

    v8f c0 = {}, c1 = {};
#pragma unroll
    for (int kb = 0; kb < D_DIM; kb += 32) {
      FragBF fa;
      fa.q[0] = *(const uint4*)&As[wm * 16 + lr][kb + hi * 8];
      fa.q[1] = *(const uint4*)&As[wm * 16 + lr][kb + 16 + hi * 8];
      const int n0 = wn * 32 + lr;
      FragBF fb0, fb1;
      fb0.q[0] = *(const uint4*)&Bs[cur][n0][kb + hi * 8];
      fb0.q[1] = *(const uint4*)&Bs[cur][n0][kb + 16 + hi * 8];
      fb1.q[0] = *(const uint4*)&Bs[cur][n0 + 16][kb + hi * 8];
      fb1.q[1] = *(const uint4*)&Bs[cur][n0 + 16][kb + 16 + hi * 8];
      c0 = __builtin_amdgcn_wmma_f32_16x16x32_bf16(false, fa.v, false, fb0.v,
                                                   (short)0, c0, false, false);
      c1 = __builtin_amdgcn_wmma_f32_16x16x32_bf16(false, fa.v, false, fb1.v,
                                                   (short)0, c1, false, false);
    }

    // ---- scores -> LDS tile ----
    const int colg0 = nc * 64 + wn * 32 + lr;
    const float n2_0 = cbn[colg0];
    const float n2_1 = cbn[colg0 + 16];
    const int rl = wm * 16 + hi * 8;
#pragma unroll
    for (int r = 0; r < 8; ++r) {
      dbuf[rl + r][wn * 32 + lr]      = n2_0 - 2.0f * c0[r];
      dbuf[rl + r][wn * 32 + lr + 16] = n2_1 - 2.0f * c1[r];
    }
    __syncthreads();   // also fences compute before next DMA reuses a buffer

    // ---- running per-row argmin (4 threads per row, 16 cols each) ----
#pragma unroll
    for (int c = 0; c < 16; ++c) {
      const float v = dbuf[srow][sc16 + c];
      const int gc = nc * 64 + sc16 + c;
      if (v < bestv) { bestv = v; besti = gc; }
    }
  }

  // ---- combine 4 partial minima per row ----
  redv[tid] = bestv;
  redi[tid] = besti;
  __syncthreads();
  if (tid < 64) {
    float bv = redv[tid * 4];
    int bi = redi[tid * 4];
#pragma unroll
    for (int j = 1; j < 4; ++j) {
      const float v = redv[tid * 4 + j];
      const int ii = redi[tid * 4 + j];
      if (v < bv || (v == bv && ii < bi)) { bv = v; bi = ii; }
    }
    const int rowg = bm + tid;
    idxbuf[rowg] = bi;
    codes[(size_t)rowg * NCB + step] = (float)bi;  // codes output
  }
}

// ---------------------------------------------------------------------------
// Gather z_q_emb = cb[idx] (f32), commitment-loss partial sum, stage GEMM3 A.
// ---------------------------------------------------------------------------
__global__ void rvq_gather_loss(const float* __restrict__ cbf,
                                const float* __restrict__ latents,
                                const int* __restrict__ idxbuf,
                                float* __restrict__ zqst,
                                float* __restrict__ acc, int step) {
  __shared__ float red[256];
  const int b = blockIdx.x, t = threadIdx.x;
  const int idx = idxbuf[b];
  const float e = cbf[(size_t)idx * D_DIM + t];
  const float ze = latents[(size_t)b * C_DIM + step * D_DIM + t];
  zqst[(size_t)b * D_DIM + t] = e;   // forward value of straight-through
  const float d = ze - e;
  red[t] = d * d;
  __syncthreads();
  for (int s = 128; s > 0; s >>= 1) {
    if (t < s) red[t] += red[t + s];
    __syncthreads();
  }
  if (t == 0) atomicAdd(acc, red[0]);   // global_atomic_add_f32
}

__global__ void rvq_finalize(const float* __restrict__ acc,
                             float* __restrict__ lossout) {
  const float enc = 0.25f * (*acc) / ((float)B_ROWS * (float)D_DIM);
  lossout[0] = enc;   // vq_loss = enc + 0
  lossout[1] = enc;   // enc_loss
  lossout[2] = 0.0f;  // cbk_loss (CODEBOOK_COST == 0)
}

// ---------------------------------------------------------------------------
extern "C" void kernel_launch(void* const* d_in, const int* in_sizes, int n_in,
                              void* d_out, int out_size, void* d_ws,
                              size_t ws_size, hipStream_t stream) {
  const float* z  = (const float*)d_in[0];
  const float* cb = (const float*)d_in[1];
  const float* Wi = (const float*)d_in[2];
  const float* bi = (const float*)d_in[3];
  const float* Wo = (const float*)d_in[4];
  const float* bo = (const float*)d_in[5];
  float* out = (float*)d_out;

  const size_t BC = (size_t)B_ROWS * C_DIM;
  float* zq      = out;                                   // [B,C]
  float* losses  = out + BC;                              // 3 scalars
  float* codes   = out + BC + 3;                          // [B,4] (as float)
  float* latents = out + BC + 3 + (size_t)B_ROWS * NCB;   // [B, 4*D]

  char* ws = (char*)d_ws;
  float*  resid = (float*)ws;  ws += BC * sizeof(float);
  float*  zqst  = (float*)ws;  ws += (size_t)B_ROWS * D_DIM * sizeof(float);
  __bf16* Wi_b  = (__bf16*)ws; ws += (size_t)NCB * C_DIM * D_DIM * sizeof(__bf16);
  __bf16* Wo_b  = (__bf16*)ws; ws += (size_t)NCB * D_DIM * C_DIM * sizeof(__bf16);
  __bf16* cb_b  = (__bf16*)ws; ws += (size_t)NCB * K_CB * D_DIM * sizeof(__bf16);
  float*  cbn   = (float*)ws;  ws += (size_t)NCB * K_CB * sizeof(float);
  int*    idxb  = (int*)ws;    ws += (size_t)B_ROWS * sizeof(int);
  float*  acc   = (float*)ws;

  rvq_init<<<(unsigned)(BC / 4 / 256), 256, 0, stream>>>(z, zq, resid, acc);

  const int nW = NCB * C_DIM * D_DIM;
  f32_to_bf16<<<(nW + 255) / 256, 256, 0, stream>>>(Wi, Wi_b, nW);
  f32_to_bf16<<<(nW + 255) / 256, 256, 0, stream>>>(Wo, Wo_b, nW);
  const int nCb = NCB * K_CB * D_DIM;
  f32_to_bf16<<<(nCb + 255) / 256, 256, 0, stream>>>(cb, cb_b, nCb);
  cb_norms<<<NCB * K_CB, 256, 0, stream>>>(cb, cbn);

  for (int s = 0; s < NCB; ++s) {
    // z_e = residual @ W_in[s] + b_in[s]  -> latents[:, s*D:(s+1)*D]
    rvq_gemm<<<dim3(D_DIM / 64, B_ROWS / 64), 256, 0, stream>>>(
        resid, C_DIM, C_DIM, Wi_b + (size_t)s * C_DIM * D_DIM, D_DIM,
        bi + s * D_DIM, latents + s * D_DIM, C_DIM, nullptr, nullptr, 0);

    // argmin_k ||z_e - cb[k]||^2  (TDM double-buffered codebook tiles)
    rvq_dist_argmin<<<B_ROWS / 64, 256, 0, stream>>>(
        latents + s * D_DIM, C_DIM, cb_b + (size_t)s * K_CB * D_DIM,
        cbn + s * K_CB, idxb, codes, s);

    // gather + commitment loss + stage A for GEMM3
    rvq_gather_loss<<<B_ROWS, 256, 0, stream>>>(
        cb + (size_t)s * K_CB * D_DIM, latents, idxb, zqst, acc, s);

    // z_q_i = cb[idx] @ W_out[s] + b_out[s];  z_q += z_q_i; residual -= z_q_i
    rvq_gemm<<<dim3(C_DIM / 64, B_ROWS / 64), 256, 0, stream>>>(
        zqst, D_DIM, D_DIM, Wo_b + (size_t)s * D_DIM * C_DIM, C_DIM,
        bo + s * C_DIM, zq, C_DIM, zq, resid, 1);
  }

  rvq_finalize<<<1, 1, 0, stream>>>(acc, losses);
}
